// CrossAttention_37512244364058
// MI455X (gfx1250) — matrix operand order repaired and checked by reading, not compiled, not run on hardware
//
#include <hip/hip_runtime.h>

typedef _Float16 v16h __attribute__((ext_vector_type(16)));
typedef _Float16 v8h  __attribute__((ext_vector_type(8)));
typedef _Float16 v4h  __attribute__((ext_vector_type(4)));
typedef float    v8f  __attribute__((ext_vector_type(8)));
typedef unsigned int u32x4 __attribute__((ext_vector_type(4)));
typedef int          i32x4 __attribute__((ext_vector_type(4)));
typedef int          i32x8 __attribute__((ext_vector_type(8)));

#define N_TOK   4096   // 64*64
#define CHN     256
#define NHEADS  8
#define HDIM    32
#define SCALE_F 0.17677669529663687f   // 1/sqrt(32)

__device__ __forceinline__ v8f wmma16x16x32(v16h a, v16h b, v8f c) {
    return __builtin_amdgcn_wmma_f32_16x16x32_f16(
        /*neg_a=*/false, a, /*neg_b=*/false, b,
        /*c_mod=*/(short)0, c, /*reuse_a=*/false, /*reuse_b=*/false);
}

// ---------------------------------------------------------------------------
// Tensor Data Mover: load a 32-row x 32-half f16 tile (row stride in elements)
// into LDS. D# per CDNA5 ISA. Falls back to plain
// per-lane copies if the builtin is unavailable.
// ---------------------------------------------------------------------------
#if defined(__has_builtin)
#if __has_builtin(__builtin_amdgcn_tensor_load_to_lds) && __has_builtin(__builtin_amdgcn_s_wait_tensorcnt)
#define HAVE_TDM 1
#endif
#endif

__device__ __forceinline__ void tdm_load_tile32x32(_Float16* lds_dst,
                                                   const _Float16* gsrc,
                                                   unsigned row_stride_elems,
                                                   int lane) {
#ifdef HAVE_TDM
    unsigned long long ga = (unsigned long long)gsrc;
    unsigned lds_off = (unsigned)(unsigned long long)(void*)lds_dst;  // low 32b = LDS byte offset
    u32x4 g0;
    g0[0] = 1u;                                           // count=1 (valid), user mode
    g0[1] = lds_off;                                      // lds_addr (bytes)
    g0[2] = (unsigned)(ga & 0xffffffffu);                 // global_addr[31:0]
    g0[3] = (unsigned)((ga >> 32) & 0x01ffffffu) | (2u << 30);  // addr[56:32], type=2
    unsigned td0 = row_stride_elems > 32u ? row_stride_elems : 32u;  // OOB-safe dims
    i32x8 g1;
    g1[0] = (int)(1u << 16);                              // data_size=1 -> 2 bytes/elem
    g1[1] = (int)((td0 & 0xffffu) << 16);                 // tensor_dim0[15:0]
    g1[2] = (int)(((td0 >> 16) & 0xffffu) | (0x4000u << 16)); // dim0[31:16], tensor_dim1[15:0]=16384
    g1[3] = (int)(32u << 16);                             // tensor_dim1[31:16]=0, tile_dim0=32
    g1[4] = (int)32u;                                     // tile_dim1=32, tile_dim2=0
    g1[5] = (int)row_stride_elems;                        // tensor_dim0_stride[31:0]
    g1[6] = 0;                                            // stride0[47:32], dim1_stride[15:0]
    g1[7] = 0;                                            // dim1_stride[47:16]
    i32x4 z4 = {0, 0, 0, 0};
#if __clang_major__ >= 23
    i32x8 z8 = {0, 0, 0, 0, 0, 0, 0, 0};
    __builtin_amdgcn_tensor_load_to_lds(g0, g1, z4, z4, z8, 0);
#else
    __builtin_amdgcn_tensor_load_to_lds(g0, g1, z4, z4, 0);
#endif
    (void)lane;
#else
    // Fallback: lane r copies one 32-half row (two 16B chunks).
    const v8h* src = (const v8h*)(gsrc + (unsigned)lane * row_stride_elems);
    v8h* dst = (v8h*)(lds_dst + lane * 32);
    dst[0] = src[0];
    dst[1] = src[1];
#endif
}

__device__ __forceinline__ void tdm_wait_all() {
#ifdef HAVE_TDM
    __builtin_amdgcn_s_wait_tensorcnt((short)0);
#endif
    asm volatile("" ::: "memory");
}

// ---------------------------------------------------------------------------
// f32 -> f16 conversion (vectorized x4)
// ---------------------------------------------------------------------------
__global__ void cvt_f32_to_f16(const float* __restrict__ in,
                               _Float16* __restrict__ out, int n4) {
    int i = blockIdx.x * blockDim.x + threadIdx.x;
    if (i < n4) {
        float4 v = ((const float4*)in)[i];
        v4h o = { (_Float16)v.x, (_Float16)v.y, (_Float16)v.z, (_Float16)v.w };
        ((v4h*)out)[i] = o;
    }
}

// ---------------------------------------------------------------------------
// WMMA GEMM:  Out[o, n] = sum_c W[o,c] * X[c,n]   (256 x 256) x (256 x 4096)
// BMODE 0: X is [c][4096] row-major (f16)
// BMODE 1: X is [h][n][c] per-head transposed (f16), h = c>>5
// SMODE 0: f16 store to outh[o*4096 + n]
// SMODE 1: f16 store transposed-per-head: outh[(o>>5)*4096*32 + n*32 + (o&31)]
// SMODE 2: f32 store + bias to outf[o*4096 + n]
// ---------------------------------------------------------------------------
template <int BMODE, int SMODE>
__global__ __launch_bounds__(256)
void gemm_wmma(const _Float16* __restrict__ W, const _Float16* __restrict__ X,
               _Float16* __restrict__ outh, float* __restrict__ outf,
               const float* __restrict__ bias) {
    const int lane = threadIdx.x & 31;
    const int wave = threadIdx.x >> 5;
    const int tile = blockIdx.x * 8 + wave;      // 4096 tiles total
    const int o0   = (tile >> 8) * 16;           // 16 o-tiles
    const int n0   = (tile & 255) * 16;          // 256 n-tiles
    const int half = lane >> 4;
    const int l15  = lane & 15;

    v8f acc = {};
    #pragma unroll
    for (int k0 = 0; k0 < CHN; k0 += 32) {
        v16h a;
        {
            const _Float16* wr = W + (o0 + l15) * CHN + k0 + half * 16;
            v8h lo = *(const v8h*)(wr);
            v8h hi = *(const v8h*)(wr + 8);
            #pragma unroll
            for (int i = 0; i < 8; ++i) { a[i] = lo[i]; a[i + 8] = hi[i]; }
        }
        v16h b;
        if (BMODE == 0) {
            const _Float16* xp = X + (k0 + half * 16) * N_TOK + n0 + l15;
            #pragma unroll
            for (int e = 0; e < 16; ++e) b[e] = xp[e * N_TOK];  // coalesced across lanes
        } else {
            const _Float16* xp =
                X + ((k0 >> 5) * N_TOK + n0 + l15) * HDIM + half * 16;
            v8h b0 = *(const v8h*)(xp);
            v8h b1 = *(const v8h*)(xp + 8);
            #pragma unroll
            for (int i = 0; i < 8; ++i) { b[i] = b0[i]; b[i + 8] = b1[i]; }
        }
        acc = wmma16x16x32(a, b, acc);
    }
    #pragma unroll
    for (int r = 0; r < 8; ++r) {
        const int o = o0 + r + half * 8;
        const int n = n0 + l15;
        if (SMODE == 0) {
            outh[o * N_TOK + n] = (_Float16)acc[r];
        } else if (SMODE == 1) {
            outh[(o >> 5) * (N_TOK * HDIM) + n * HDIM + (o & 31)] = (_Float16)acc[r];
        } else {
            outf[o * N_TOK + n] = acc[r] + bias[o];
        }
    }
}

// ---------------------------------------------------------------------------
// Flash attention, one wave per (head, 16-query block).
// Qt, Kt: [h][n][c] f16 (c contiguous).  V: [o][N] f16.  Fused out: [h][n][c].
// K/V 32x32 tiles stream through LDS via TDM (double-buffered, s_wait_tensorcnt).
// S = Q^T K (2 WMMAs), online softmax (max via shfl, sum via P x ones WMMA),
// O += P * V^T (2 WMMAs).  5 WMMAs per 32-key tile.
// ---------------------------------------------------------------------------
__global__ __launch_bounds__(32)
void attn_flash(const _Float16* __restrict__ Qt, const _Float16* __restrict__ Kt,
                const _Float16* __restrict__ V,  _Float16* __restrict__ Fused) {
    __shared__ _Float16 kbuf[2][32 * 32];   // [m][c] tile, rows 64B
    __shared__ _Float16 vbuf[2][32 * 32];   // [c][m] tile, rows 64B
    __shared__ _Float16 pls[16 * 32];       // P tile [n][m]
    const int lane = threadIdx.x & 31;
    const int half = lane >> 4;
    const int l15  = lane & 15;
    const int h    = blockIdx.x >> 8;            // 8 heads
    const int n0   = (blockIdx.x & 255) * 16;    // 256 query blocks

    const _Float16* ktH   = Kt + h * (N_TOK * HDIM);   // [n][c]
    const _Float16* vbase = V + (h * HDIM) * N_TOK;    // [c][N]

    // Q A-fragment: row n = l15, k = c
    v16h aq;
    {
        const _Float16* qp = Qt + (h * N_TOK + n0 + l15) * HDIM + half * 16;
        v8h lo = *(const v8h*)(qp);
        v8h hi = *(const v8h*)(qp + 8);
        #pragma unroll
        for (int i = 0; i < 8; ++i) { aq[i] = lo[i]; aq[i + 8] = hi[i]; }
    }

    // All-ones B fragment: P x ones computes row sums of P in every D column.
    v16h bones;
    #pragma unroll
    for (int i = 0; i < 16; ++i) bones[i] = (_Float16)1.0f;

    float run_max[8], run_sum[8], alpha[8];
    v8f oc0 = {}, oc1 = {};
    #pragma unroll
    for (int r = 0; r < 8; ++r) { run_max[r] = -1e30f; run_sum[r] = 0.0f; }

    // Prologue: fetch first K/V tiles into buffer 0.
    tdm_load_tile32x32(kbuf[0], ktH, HDIM, lane);          // rows m0..31 of [n][c]
    tdm_load_tile32x32(vbuf[0], vbase, N_TOK, lane);       // rows c0..31 of [c][N]
    tdm_wait_all();

    for (int m0 = 0; m0 < N_TOK; m0 += 32) {
        const int cur = (m0 >> 5) & 1;
        if (m0 + 32 < N_TOK) {   // prefetch next tiles into the other buffer
            tdm_load_tile32x32(kbuf[cur ^ 1], ktH + (m0 + 32) * HDIM, HDIM, lane);
            tdm_load_tile32x32(vbuf[cur ^ 1], vbase + (m0 + 32), N_TOK, lane);
        }

        // K B-fragments from LDS: col m = l15 (+16), element e -> c = 16*half + e
        v16h bk0, bk1;
        {
            const _Float16* kp0 = kbuf[cur] + l15 * 32 + half * 16;
            const _Float16* kp1 = kp0 + 16 * 32;
            v8h x0 = *(const v8h*)kp0, x1 = *(const v8h*)(kp0 + 8);
            v8h y0 = *(const v8h*)kp1, y1 = *(const v8h*)(kp1 + 8);
            #pragma unroll
            for (int i = 0; i < 8; ++i) {
                bk0[i] = x0[i]; bk0[i + 8] = x1[i];
                bk1[i] = y0[i]; bk1[i + 8] = y1[i];
            }
        }
        v8f sa = {}, sb = {};
        sa = wmma16x16x32(aq, bk0, sa);          // keys m0 .. m0+15
        sb = wmma16x16x32(aq, bk1, sb);          // keys m0+16 .. m0+31

        // Online softmax: row n = r + 8*half lives in the 16 lanes of this half
        #pragma unroll
        for (int r = 0; r < 8; ++r) {
            float s0 = sa[r] * SCALE_F, s1 = sb[r] * SCALE_F;
            float mx = fmaxf(s0, s1);
            mx = fmaxf(mx, __shfl_xor(mx, 1, 32));
            mx = fmaxf(mx, __shfl_xor(mx, 2, 32));
            mx = fmaxf(mx, __shfl_xor(mx, 4, 32));
            mx = fmaxf(mx, __shfl_xor(mx, 8, 32));
            float nm = fmaxf(run_max[r], mx);
            alpha[r] = __expf(run_max[r] - nm);
            run_max[r] = nm;
            float p0 = __expf(s0 - nm), p1 = __expf(s1 - nm);
            oc0[r] *= alpha[r]; oc1[r] *= alpha[r];
            const int nn = (r + half * 8) * 32;
            pls[nn + l15]      = (_Float16)p0;
            pls[nn + l15 + 16] = (_Float16)p1;
        }
        __syncthreads();  // single-wave WG: cheap; orders pls stores vs. loads

        // P A-fragment from LDS
        v16h ap;
        {
            const _Float16* pp = pls + l15 * 32 + half * 16;
            v8h lo = *(const v8h*)pp;
            v8h hi = *(const v8h*)(pp + 8);
            #pragma unroll
            for (int i = 0; i < 8; ++i) { ap[i] = lo[i]; ap[i + 8] = hi[i]; }
        }
        // Row sums of P via WMMA against all-ones B
        v8f sums = {};
        sums = wmma16x16x32(ap, bones, sums);
        #pragma unroll
        for (int r = 0; r < 8; ++r) run_sum[r] = run_sum[r] * alpha[r] + sums[r];

        // V^T B-fragments from LDS: col c = l15 (+16), element e -> m = 16*half + e
        v16h bv0, bv1;
        {
            const _Float16* vp0 = vbuf[cur] + l15 * 32 + half * 16;
            const _Float16* vp1 = vp0 + 16 * 32;
            v8h x0 = *(const v8h*)vp0, x1 = *(const v8h*)(vp0 + 8);
            v8h y0 = *(const v8h*)vp1, y1 = *(const v8h*)(vp1 + 8);
            #pragma unroll
            for (int i = 0; i < 8; ++i) {
                bv0[i] = x0[i]; bv0[i + 8] = x1[i];
                bv1[i] = y0[i]; bv1[i + 8] = y1[i];
            }
        }
        oc0 = wmma16x16x32(ap, bv0, oc0);
        oc1 = wmma16x16x32(ap, bv1, oc1);

        __syncthreads();      // done reading pls before next iteration rewrites it
        tdm_wait_all();       // next-iteration K/V tiles have landed in LDS
    }

    // O D-layout: row n = r + 8*half, col c = l15 (tile0) / 16+l15 (tile1)
    #pragma unroll
    for (int r = 0; r < 8; ++r) {
        float inv = 1.0f / run_sum[r];
        const int nn = n0 + r + half * 8;
        Fused[(h * N_TOK + nn) * HDIM + l15]      = (_Float16)(oc0[r] * inv);
        Fused[(h * N_TOK + nn) * HDIM + 16 + l15] = (_Float16)(oc1[r] * inv);
    }
}

// ---------------------------------------------------------------------------
// Launch
// ---------------------------------------------------------------------------
extern "C" void kernel_launch(void* const* d_in, const int* in_sizes, int n_in,
                              void* d_out, int out_size, void* d_ws, size_t ws_size,
                              hipStream_t stream) {
    const float* x1 = (const float*)d_in[0];
    const float* x2 = (const float*)d_in[1];
    const float* Wq = (const float*)d_in[2];
    const float* Wk = (const float*)d_in[3];
    const float* Wv = (const float*)d_in[4];
    const float* Wp = (const float*)d_in[5];
    const float* bp = (const float*)d_in[6];

    const int NX = CHN * N_TOK;   // 1048576
    const int NW = CHN * CHN;     // 65536

    _Float16* ws  = (_Float16*)d_ws;
    _Float16* XH1 = ws;
    _Float16* XH2 = XH1 + NX;
    _Float16* WQh = XH2 + NX;
    _Float16* WKh = WQh + NW;
    _Float16* WVh = WKh + NW;
    _Float16* WPh = WVh + NW;
    _Float16* QT  = WPh + NW;     // [h][n][c]
    _Float16* KT  = QT + NX;      // [h][n][c]
    _Float16* VN  = KT + NX;      // [o][N]
    _Float16* FU  = VN + NX;      // [h][n][c]

    cvt_f32_to_f16<<<(NX / 4 + 255) / 256, 256, 0, stream>>>(x1, XH1, NX / 4);
    cvt_f32_to_f16<<<(NX / 4 + 255) / 256, 256, 0, stream>>>(x2, XH2, NX / 4);
    cvt_f32_to_f16<<<(NW / 4 + 255) / 256, 256, 0, stream>>>(Wq, WQh, NW / 4);
    cvt_f32_to_f16<<<(NW / 4 + 255) / 256, 256, 0, stream>>>(Wk, WKh, NW / 4);
    cvt_f32_to_f16<<<(NW / 4 + 255) / 256, 256, 0, stream>>>(Wv, WVh, NW / 4);
    cvt_f32_to_f16<<<(NW / 4 + 255) / 256, 256, 0, stream>>>(Wp, WPh, NW / 4);

    gemm_wmma<0, 1><<<512, 256, 0, stream>>>(WQh, XH1, QT, nullptr, nullptr);
    gemm_wmma<0, 1><<<512, 256, 0, stream>>>(WKh, XH2, KT, nullptr, nullptr);
    gemm_wmma<0, 0><<<512, 256, 0, stream>>>(WVh, XH2, VN, nullptr, nullptr);

    attn_flash<<<NHEADS * 256, 32, 0, stream>>>(QT, KT, VN, FU);

    gemm_wmma<1, 2><<<512, 256, 0, stream>>>(WPh, FU, nullptr, (float*)d_out, bp);
}
